// SymModel_41085657154025
// MI455X (gfx1250) — matrix operand order repaired and checked by reading, not compiled, not run on hardware
//
#include <hip/hip_runtime.h>
#include <math.h>

// Problem constants (match reference)
#define S_LEN 128
#define B_SZ  256
#define V_SZ  256
#define NC    32
#define RR    8
#define C_SZ  (NC*RR)   // 256

typedef __attribute__((ext_vector_type(16))) _Float16 v16h;
typedef __attribute__((ext_vector_type(8)))  _Float16 v8h;
typedef __attribute__((ext_vector_type(8)))  float    v8f;

// ---------------------------------------------------------------------------
// WMMA fragment loaders (wave32, v_wmma_f32_16x16x32_f16, contiguous k order shared by A and B)
// A (16x32 f16), row-major [M][K] with leading dim ldk:
//   lane: M=lane&15, K = k0 + 16*(lane>=16) + j (matches B fragment order)
__device__ __forceinline__ v16h load_a_frag(const _Float16* base, int ldk, int k0, int lane) {
    int M  = lane & 15;
    int hi = (lane >> 4) & 1;
    return *(const v16h*)(base + M * ldk + k0 + hi * 16);
}
// B (32x16 f16): D[M,N] = sum_K A[M,K]*W[n0+N][K], i.e. B[K][N]=W[n0+N][K],
// W stored row-major [N][K] with leading dim ldk. lane: N=lane&15, K = k0 + 16*(lane>=16) + j.
__device__ __forceinline__ v16h load_b_frag(const _Float16* wbase, int ldk, int n0, int k0, int lane) {
    int N  = lane & 15;
    int hi = (lane >> 4) & 1;
    return *(const v16h*)(wbase + (size_t)(n0 + N) * ldk + k0 + hi * 16);
}

__device__ __forceinline__ float sigmoidf_(float x) { return 1.0f / (1.0f + __expf(-x)); }
__device__ __forceinline__ float lrelu_(float x)    { return x > 0.0f ? x : 0.01f * x; }

// ---------------------------------------------------------------------------
// Phase 1: fully parallel over (s, batch-tile). Computes cosines, var_choice,
// val, and the val-side interp factor fac[s][b][c]. One WG = one s and 16
// batch rows; 256 threads (8 waves).
__global__ __launch_bounds__(256) void phase1_val_side(
    const float* __restrict__ av_, const float* __restrict__ xn_,
    const float* __restrict__ xv_, const float* __restrict__ yn_,
    const float* __restrict__ yv_, const float* __restrict__ sw,   // [C][2V]
    const float* __restrict__ snw,                                  // [C][2]
    const float* __restrict__ defv,                                 // [4][V]
    const float* __restrict__ vnw,                                  // [6][3]
    float* __restrict__ fac)                                        // [S][B][C]
{
    __shared__ __align__(32) _Float16 Wq1h[C_SZ * V_SZ];  // 128 KB
    __shared__ __align__(32) _Float16 valh[16 * V_SZ];    // 8 KB
    __shared__ float invnv[16];
    __shared__ float invW1[C_SZ];
    __shared__ float s1s[C_SZ];

    const int tid = threadIdx.x;
    const int b0  = blockIdx.x * 16;
    const int s   = blockIdx.y;
    const float eps = 1e-8f;

    // Coalesced cooperative copy: Wq[:,1,:] f32 global -> f16 LDS.
    // Adjacent lanes read adjacent addresses within a row.
    for (int i = tid; i < C_SZ * V_SZ; i += 256) {
        int c = i >> 8, k = i & 255;
        Wq1h[i] = (_Float16)sw[(size_t)c * 512 + 256 + k];
    }
    __syncthreads();
    // Per-row inverse norm (from LDS) + sigmoid(nand[:,1]); thread tid = row c.
    {
        int c = tid;
        float n = 0.0f;
        for (int k = 0; k < V_SZ; ++k) {
            float w = (float)Wq1h[c * V_SZ + k];
            n += w * w;
        }
        invW1[c] = 1.0f / fmaxf(sqrtf(n), eps);
        s1s[c]   = sigmoidf_(snw[c * 2 + 1]);
    }

    // Cosines + var_choice + val. Thread layout: row = tid/16, 16 threads per row.
    const int row = tid >> 4, l16 = tid & 15;
    const size_t base = ((size_t)s * B_SZ + (b0 + row)) * V_SZ;
    float dax = 0, day = 0, dxy = 0, na = 0, nx = 0, ny = 0;
    float xvv[16], yvv[16];
#pragma unroll
    for (int k = 0; k < 16; ++k) {
        int v = l16 + k * 16;
        float a = av_[base + v], xn = xn_[base + v], yn = yn_[base + v];
        xvv[k] = xv_[base + v];  yvv[k] = yv_[base + v];
        dax += a * xn; day += a * yn; dxy += xn * yn;
        na += a * a;   nx += xn * xn; ny += yn * yn;
    }
#pragma unroll
    for (int m = 8; m >= 1; m >>= 1) {
        dax += __shfl_xor(dax, m, 16); day += __shfl_xor(day, m, 16);
        dxy += __shfl_xor(dxy, m, 16); na  += __shfl_xor(na,  m, 16);
        nx  += __shfl_xor(nx,  m, 16); ny  += __shfl_xor(ny,  m, 16);
    }
    float sna = fmaxf(sqrtf(na), eps), snx = fmaxf(sqrtf(nx), eps), sny = fmaxf(sqrtf(ny), eps);
    float cs3[3];
    cs3[0] = dax / (sna * snx);
    cs3[1] = day / (sna * sny);
    cs3[2] = dxy / (snx * sny);

    float vc[6];
#pragma unroll
    for (int i = 0; i < 6; ++i) {
        float p = 1.0f;
#pragma unroll
        for (int j = 0; j < 3; ++j) {
            float nw = sigmoidf_(vnw[i * 3 + j]);
            p *= nw * cs3[j] + (1.0f - nw) * (1.0f - cs3[j]);
        }
        vc[i] = p;
    }

    float nv = 0.0f;
#pragma unroll
    for (int k = 0; k < 16; ++k) {
        int v = l16 + k * 16;
        float val = vc[0] * xvv[k] + vc[1] * yvv[k];
#pragma unroll
        for (int d = 0; d < 4; ++d) val += vc[2 + d] * defv[d * V_SZ + v];
        valh[row * V_SZ + v] = (_Float16)val;
        nv += val * val;
    }
#pragma unroll
    for (int m = 8; m >= 1; m >>= 1) nv += __shfl_xor(nv, m, 16);
    if (l16 == 0) invnv[row] = 1.0f / fmaxf(sqrtf(nv), eps);
    __syncthreads();

    // GEMM: dots[16, C] = val[16,V] x Wq1^T, then normalize/lrelu/interp -> fac
    const int wave = tid >> 5, lane = tid & 31;
    const int N = lane & 15, mh = ((lane >> 4) & 1) * 8;
#pragma unroll
    for (int t = 0; t < 2; ++t) {
        int c0 = (wave * 2 + t) * 16;
        v8f acc = {};
#pragma unroll
        for (int kk = 0; kk < 8; ++kk) {
            v16h aF = load_a_frag(valh, V_SZ, kk * 32, lane);
            v16h bF = load_b_frag(Wq1h, V_SZ, c0, kk * 32, lane);
            acc = __builtin_amdgcn_wmma_f32_16x16x32_f16(false, aF, false, bF,
                                                         (short)0, acc, false, false);
        }
        int c = c0 + N;
        float iw = invW1[c], s1 = s1s[c];
#pragma unroll
        for (int r = 0; r < 8; ++r) {
            int M = r + mh;
            float d = acc[r] * invnv[M] * iw;
            float g = lrelu_(d);
            fac[((size_t)s * B_SZ + (b0 + M)) * C_SZ + c] = s1 * g + (1.0f - s1) * (1.0f - g);
        }
    }
}

// ---------------------------------------------------------------------------
// Phase 2: sequential scan. 16 persistent WGs, each owning 16 batch rows.
// Wq[:,0,:] (f16, 128 KB) + summer_vecs^T live in LDS for all 128 steps.
__global__ __launch_bounds__(256) void phase2_scan(
    const float* __restrict__ sw,   // [C][2V]
    const float* __restrict__ snw,  // [C][2]
    const float* __restrict__ sv,   // [NC][V]
    const float* __restrict__ fac,  // [S][B][C]
    float* __restrict__ out)        // [B][S][V]
{
    __shared__ __align__(32) _Float16 Wq0h[C_SZ * V_SZ];  // 128 KB
    __shared__ __align__(32) _Float16 svT[V_SZ * NC];     // 16 KB  (svT[v][n])
    __shared__ __align__(32) _Float16 rsh[16 * V_SZ];     // 8 KB   running_sum f16
    __shared__ float ch[16 * C_SZ];                       // 16 KB  choices
    __shared__ __align__(32) _Float16 wAh[16 * NC];       // 1 KB   R-reduced weights
    __shared__ float invW0[C_SZ], s0s[C_SZ];
    __shared__ float nrm[2][16];                          // |rs|^2 double buffer

    const int tid = threadIdx.x;
    const int b0  = blockIdx.x * 16;
    const float eps = 1e-8f;

    // Coalesced cooperative copy: Wq[:,0,:] f32 global -> f16 LDS.
    for (int i = tid; i < C_SZ * V_SZ; i += 256) {
        int c = i >> 8, k = i & 255;
        Wq0h[i] = (_Float16)sw[(size_t)c * 512 + k];
    }
    {   // summer_vecs transposed -> LDS f16 (coalesced across lanes)
        int v = tid;
        for (int n = 0; n < NC; ++n) svT[v * NC + n] = (_Float16)sv[n * V_SZ + v];
    }
    for (int i = tid; i < 16 * V_SZ; i += 256) rsh[i] = (_Float16)1e-6f;
    if (tid < 16) { nrm[0][tid] = 256.0f * 1e-12f; nrm[1][tid] = 0.0f; }
    __syncthreads();
    {   // per-row inverse norm (from LDS) + sigmoid(nand[:,0])
        int c = tid;
        float n = 0.0f;
        for (int k = 0; k < V_SZ; ++k) {
            float w = (float)Wq0h[c * V_SZ + k];
            n += w * w;
        }
        invW0[c] = 1.0f / fmaxf(sqrtf(n), eps);
        s0s[c]   = sigmoidf_(snw[c * 2 + 0]);
    }
    __syncthreads();

    const int wave = tid >> 5, lane = tid & 31;
    const int N = lane & 15, mh = ((lane >> 4) & 1) * 8;

    for (int s = 0; s < S_LEN; ++s) {
        const int cur = s & 1, nxt = cur ^ 1;

        // Prefetch next step's fac tile (16 rows x 256 c = 16 KB) toward WGP
        // while this step's WMMAs run (scan is latency-bound).
        if (s + 1 < S_LEN) {
            const float* pf = &fac[((size_t)(s + 1) * B_SZ + b0) * C_SZ];
            __builtin_prefetch(pf + tid * 16, 0, 0);
        }

        // GEMM1: dots = rs[16,V] x Wq0^T -> normalize/lrelu/interp * fac -> ch
#pragma unroll
        for (int t = 0; t < 2; ++t) {
            int c0 = (wave * 2 + t) * 16;
            v8f acc = {};
#pragma unroll
            for (int kk = 0; kk < 8; ++kk) {
                v16h aF = load_a_frag(rsh, V_SZ, kk * 32, lane);
                v16h bF = load_b_frag(Wq0h, V_SZ, c0, kk * 32, lane);
                acc = __builtin_amdgcn_wmma_f32_16x16x32_f16(false, aF, false, bF,
                                                             (short)0, acc, false, false);
            }
            int c = c0 + N;
            float iw = invW0[c], s0 = s0s[c];
#pragma unroll
            for (int r = 0; r < 8; ++r) {
                int M = r + mh;
                float inn = 1.0f / fmaxf(sqrtf(nrm[cur][M]), eps);
                float d = acc[r] * inn * iw;
                float g = lrelu_(d);
                float p0 = s0 * g + (1.0f - s0) * (1.0f - g);
                float f = fac[((size_t)s * B_SZ + (b0 + M)) * C_SZ + c];
                ch[M * C_SZ + c] = p0 * f;
            }
        }
        __syncthreads();

        // Reduce over redundancy R=8 -> wA[16][32] f16; zero next norm buffer
        for (int i = tid; i < 16 * NC; i += 256) {
            int m = i >> 5, nc = i & 31;
            float a = 0.0f;
#pragma unroll
            for (int r = 0; r < RR; ++r) a += ch[m * C_SZ + nc * RR + r];
            wAh[m * NC + nc] = (_Float16)a;
        }
        if (tid < 16) nrm[nxt][tid] = 0.0f;
        __syncthreads();

        // GEMM2: new_rs[16,V] = wA[16,32] x summer_vecs[32,V]; K=32 = one WMMA
#pragma unroll
        for (int t = 0; t < 2; ++t) {
            int v0 = (wave * 2 + t) * 16;
            v16h aF = load_a_frag(wAh, NC, 0, lane);
            v16h bF = *(const v16h*)(&svT[(size_t)(v0 + N) * NC + ((lane >> 4) & 1) * 16]);
            v8f acc = {};
            acc = __builtin_amdgcn_wmma_f32_16x16x32_f16(false, aF, false, bF,
                                                         (short)0, acc, false, false);
            int v = v0 + N;
#pragma unroll
            for (int r = 0; r < 8; ++r) {
                int M = r + mh;
                float val = acc[r];
                out[(((size_t)(b0 + M)) * S_LEN + s) * V_SZ + v] = val;
                rsh[M * V_SZ + v] = (_Float16)val;
                atomicAdd(&nrm[nxt][M], val * val);
            }
        }
        __syncthreads();
    }
}

// ---------------------------------------------------------------------------
extern "C" void kernel_launch(void* const* d_in, const int* in_sizes, int n_in,
                              void* d_out, int out_size, void* d_ws, size_t ws_size,
                              hipStream_t stream) {
    (void)in_sizes; (void)n_in; (void)out_size; (void)ws_size;
    const float* av   = (const float*)d_in[0];
    const float* xn   = (const float*)d_in[1];
    const float* xv   = (const float*)d_in[2];
    const float* yn   = (const float*)d_in[3];
    const float* yv   = (const float*)d_in[4];
    const float* sv   = (const float*)d_in[5];
    const float* sw   = (const float*)d_in[6];
    const float* snw  = (const float*)d_in[7];
    const float* defv = (const float*)d_in[8];
    const float* vnw  = (const float*)d_in[9];
    float* fac = (float*)d_ws;           // [S][B][C] f32 = 33.5 MB
    float* out = (float*)d_out;          // [B][S][V] f32

    dim3 g1(B_SZ / 16, S_LEN);           // 2048 WGs, fully parallel
    phase1_val_side<<<g1, 256, 0, stream>>>(av, xn, xv, yn, yv, sw, snw, defv, vnw, fac);
    phase2_scan<<<dim3(B_SZ / 16), 256, 0, stream>>>(sw, snw, sv, fac, out);
}
